// Model_22265110462493
// MI455X (gfx1250) — 8-bit matrix operand order repaired and checked by reading, compiled, not run on hardware
//
#include <hip/hip_runtime.h>
#include <hip/hip_bf16.h>

typedef __attribute__((ext_vector_type(8))) int v8i;

constexpr int B_ = 8, M_ = 4096, K_ = 2048, N_ = 2048;
constexpr int MT = 128;
constexpr int NT = 128;
constexpr int KC = 64;
constexpr int LDA = 20;
constexpr int LDB = 20;
constexpr int ITERS = K_ / KC;

__device__ __forceinline__ int pack4(int a, int b, int c, int d) {
    const unsigned t0 = __builtin_amdgcn_perm((unsigned)b, (unsigned)a, 0x0c0c0400u);
    const unsigned t1 = __builtin_amdgcn_perm((unsigned)d, (unsigned)c, 0x04000c0cu);
    return (int)(t0 | t1);
}

__global__ __launch_bounds__(256) void qmm_i8_wmma(
    const int* __restrict__ x1, const int* __restrict__ x2,
    const float* __restrict__ scale, const float* __restrict__ offs,
    const float* __restrict__ bias, float* __restrict__ out)
{
    __shared__ int As[2][MT * LDA];
    __shared__ int Bs[2][NT * LDB];

    const int b  = blockIdx.z;
    const int m0 = blockIdx.y * MT;
    const int n0 = blockIdx.x * NT;

    const int tid   = threadIdx.x;
    const int lane  = tid & 31;
    const int wave  = tid >> 5;
    const int wm    = wave >> 1;
    const int wn    = wave & 1;
    const int l15   = lane & 15;
    const int khalf = lane >> 4;

    const int* __restrict__ Ap = x1 + (size_t)b * M_ * K_ + (size_t)m0 * K_;
    const int* __restrict__ Bp = x2 + (size_t)b * K_ * N_ + n0;

    const int arow = tid >> 4;
    const int akg  = tid & 15;
    const int bng  = tid & 31;
    const int bkq  = tid >> 5;

    int4 areg[8];
    int4 breg[8];

    auto gload = [&](int k0) {
        #pragma unroll
        for (int i = 0; i < 8; ++i)
            areg[i] = *(const int4*)(Ap + (size_t)(arow + i * 16) * K_ + k0 + akg * 4);
        #pragma unroll
        for (int q = 0; q < 2; ++q)
            #pragma unroll
            for (int j = 0; j < 4; ++j)
                breg[q * 4 + j] =
                    *(const int4*)(Bp + (size_t)(k0 + (bkq * 2 + q) * 4 + j) * N_ + bng * 4);
    };

    auto stage = [&](int buf) {
        #pragma unroll
        for (int i = 0; i < 8; ++i) {
            const int4 v = areg[i];
            As[buf][(arow + i * 16) * LDA + akg] = pack4(v.x, v.y, v.z, v.w);
        }
        #pragma unroll
        for (int q = 0; q < 2; ++q) {
            const int kg = bkq * 2 + q;
            const int4 r0 = breg[q * 4 + 0], r1 = breg[q * 4 + 1];
            const int4 r2 = breg[q * 4 + 2], r3 = breg[q * 4 + 3];
            Bs[buf][(bng * 4 + 0) * LDB + kg] = pack4(r0.x, r1.x, r2.x, r3.x);
            Bs[buf][(bng * 4 + 1) * LDB + kg] = pack4(r0.y, r1.y, r2.y, r3.y);
            Bs[buf][(bng * 4 + 2) * LDB + kg] = pack4(r0.z, r1.z, r2.z, r3.z);
            Bs[buf][(bng * 4 + 3) * LDB + kg] = pack4(r0.w, r1.w, r2.w, r3.w);
        }
    };

    v8i acc[2][4] = {};

    gload(0);
    stage(0);
    __syncthreads();

    for (int it = 0; it < ITERS; ++it) {
        const int cur = it & 1;
        if (it + 1 < ITERS) gload((it + 1) * KC);

        v8i a[2], bf[4];
        #pragma unroll
        for (int mt = 0; mt < 2; ++mt) {
            const int row = wm * 32 + mt * 16 + l15;
            #pragma unroll
            for (int i = 0; i < 2; ++i) {
                const int4 d = *(const int4*)&As[cur][row * LDA + i * 8 + khalf * 4];
                a[mt][4 * i] = d.x; a[mt][4 * i + 1] = d.y; a[mt][4 * i + 2] = d.z; a[mt][4 * i + 3] = d.w;
            }
        }
        #pragma unroll
        for (int nt = 0; nt < 4; ++nt) {
            const int col = wn * 64 + nt * 16 + l15;
            #pragma unroll
            for (int h = 0; h < 2; ++h) {
                const int4 d = *(const int4*)&Bs[cur][col * LDB + h * 8 + khalf * 4];
                bf[nt][4 * h] = d.x; bf[nt][4 * h + 1] = d.y;
                bf[nt][4 * h + 2] = d.z; bf[nt][4 * h + 3] = d.w;
            }
        }
        #pragma unroll
        for (int mt = 0; mt < 2; ++mt)
            #pragma unroll
            for (int nt = 0; nt < 4; ++nt)
                acc[mt][nt] = __builtin_amdgcn_wmma_i32_16x16x64_iu8(
                    false, a[mt], false, bf[nt], acc[mt][nt], false, false);

        if (it + 1 < ITERS) stage(cur ^ 1);
        __syncthreads();
    }

    const size_t outB = (size_t)b * (size_t)M_ * N_;
    #pragma unroll
    for (int mt = 0; mt < 2; ++mt) {
        #pragma unroll
        for (int nt = 0; nt < 4; ++nt) {
            const int n  = n0 + wn * 64 + nt * 16 + l15;
            const float s = scale[n];
            const float o = offs[n];
            const float bi = bias[n];
            const int mb = m0 + wm * 32 + mt * 16 + khalf * 8;
            #pragma unroll
            for (int g = 0; g < 8; ++g)
                out[outB + (size_t)(mb + g) * N_ + n] = ((float)acc[mt][nt][g] + bi) * s + o;
        }
    }
}

extern "C" void kernel_launch(void* const* d_in, const int* in_sizes, int n_in,
                              void* d_out, int out_size, void* d_ws, size_t ws_size,
                              hipStream_t stream) {
    const int*   x1     = (const int*)d_in[0];
    const int*   x2     = (const int*)d_in[1];
    const float* scale  = (const float*)d_in[2];
    const float* offset = (const float*)d_in[3];
    const float* bias   = (const float*)d_in[4];
    float*       out    = (float*)d_out;

    dim3 grid(N_ / NT, M_ / MT, B_);
    qmm_i8_wmma<<<grid, dim3(256), 0, stream>>>(x1, x2, scale, offset, bias, out);
}
